// SmallEmbedFFNN_3908420240142
// MI455X (gfx1250) — matrix operand order repaired and checked by reading, not compiled, not run on hardware
//
#include <hip/hip_runtime.h>
#include <hip/hip_bf16.h>

typedef __bf16 bf16_t;
typedef bf16_t v16bf __attribute__((ext_vector_type(16)));
typedef bf16_t v8bf  __attribute__((ext_vector_type(8)));
typedef float  v8f   __attribute__((ext_vector_type(8)));

#define B_SZ   2048
#define P_SZ   20
#define D_SZ   64
#define V_SZ   100000
#define T_SZ   33
#define FIN    2240      // 35 * 64
#define H1     512
#define H2     256
#define NROWS  (B_SZ * P_SZ)          // 40960
#define NKT1   (FIN / 32)             // 70 K-tiles for GEMM1
#define NNT1   (H1 / 16)              // 32 N-tiles for GEMM1
#define NKT2   (H1 / 32)              // 16 K-tiles for GEMM2
#define NNT2   (H2 / 16)              // 16 N-tiles for GEMM2

// workspace layout (bytes)
#define OFF_W1P    ((size_t)0)                       // 70*32*512 bf16 = 2,293,760 B
#define OFF_W2P    ((size_t)2293760)                 // 16*16*512 bf16 =   262,144 B
#define OFF_LOGITS ((size_t)2555904)                 // 40960 f32      =   163,840 B

static __device__ __forceinline__ v16bf cat16(v8bf lo, v8bf hi) {
  return __builtin_shufflevector(lo, hi, 0,1,2,3,4,5,6,7,8,9,10,11,12,13,14,15);
}

// ---------------------------------------------------------------------------
// Pack a row-major (K x N) f32 weight matrix into bf16 WMMA B-fragments.
// Fragment layout (per 32xK-step x 16-N tile): lane l owns column n = l&15,
// K-values 8*(l>>4)+e for e<8 and 16+8*(l>>4)+(e-8) for e>=8 (32 B per lane).
// ---------------------------------------------------------------------------
__global__ void pack_w_kernel(const float* __restrict__ W, bf16_t* __restrict__ Wp,
                              int N, int nNt, int total) {
  int i = blockIdx.x * blockDim.x + threadIdx.x;
  if (i >= total) return;
  int e    = i & 15;
  int lane = (i >> 4) & 31;
  int frag = i >> 9;
  int nt   = frag % nNt;
  int kt   = frag / nNt;
  int n = nt * 16 + (lane & 15);
  int k = kt * 32 + ((e >> 3) << 4) + ((lane >> 4) << 3) + (e & 7);
  Wp[i] = (bf16_t)W[(size_t)k * N + n];
}

// ---------------------------------------------------------------------------
// Fused gather + 3-layer MLP for one 16-row tile. 256 threads = 8 wave32.
// ---------------------------------------------------------------------------
__global__ __launch_bounds__(256)
void ffnn_kernel(const float* __restrict__ x_raw,
                 const int*   __restrict__ x_idx,
                 const float* __restrict__ emb,
                 const bf16_t* __restrict__ W1p,
                 const float* __restrict__ b1,
                 const bf16_t* __restrict__ W2p,
                 const float* __restrict__ b2,
                 const float* __restrict__ W3,
                 const float* __restrict__ b3,
                 float* __restrict__ logits) {
  __shared__ alignas(16) bf16_t featbuf[16][D_SZ];   //  2 KB
  __shared__ alignas(16) bf16_t h1buf[16][H1];       // 16 KB
  __shared__ alignas(16) bf16_t h2buf[16][H2];       //  8 KB

  const int tid   = threadIdx.x;
  const int wave  = tid >> 5;
  const int lane  = tid & 31;
  const int m     = lane & 15;            // A-matrix row / C column index
  const int ksel  = (lane >> 4) << 3;     // A-fragment K half-select: 0 or 8
  const int mbase = (lane >> 4) << 3;     // C/D row base: 0 or 8
  const int rg    = tid >> 4;             // gather row   (0..15)
  const int cg    = (tid & 15) << 2;      // gather col*4 (0..60)
  const int rowbase = blockIdx.x << 4;
  const int gr      = rowbase + rg;       // global feature row (= b*20+p)

  // ---------------- GEMM1: feat(16x2240) @ W1(2240x512), fused gather -------
  v8f acc1[4] = {};
  for (int j = 0; j < 35; ++j) {
    float4 vals;
    if (j < 2) {                                   // broadcast x_raw scalar
      float x = x_raw[gr * 2 + j];
      vals = make_float4(x, x, x, x);
    } else {                                       // embedding gather
      int t  = j - 2;
      int id = x_idx[gr * T_SZ + t];
      vals = *(const float4*)(emb + ((size_t)t * V_SZ + (size_t)id) * D_SZ + cg);
    }
    __syncthreads();                               // prior K-step reads done
    featbuf[rg][cg + 0] = (bf16_t)vals.x;
    featbuf[rg][cg + 1] = (bf16_t)vals.y;
    featbuf[rg][cg + 2] = (bf16_t)vals.z;
    featbuf[rg][cg + 3] = (bf16_t)vals.w;
    __syncthreads();
#pragma unroll
    for (int kk = 0; kk < 2; ++kk) {               // two K=32 steps per block
      v16bf a = cat16(*(const v8bf*)&featbuf[m][kk * 32 + ksel],
                      *(const v8bf*)&featbuf[m][kk * 32 + ksel + 16]);
      const int kt = j * 2 + kk;
#pragma unroll
      for (int i = 0; i < 4; ++i) {                // 4 N-tiles per wave
        const int nt = wave * 4 + i;
        const bf16_t* bp = W1p + (((size_t)kt * NNT1 + nt) * 32 + lane) * 16;
        v16bf bb = cat16(*(const v8bf*)bp, *(const v8bf*)(bp + 8));
        acc1[i] = __builtin_amdgcn_wmma_f32_16x16x32_bf16(
            false, a, false, bb, (short)0, acc1[i], false, false);
      }
    }
  }

  __syncthreads();
#pragma unroll
  for (int i = 0; i < 4; ++i) {                    // bias + ReLU -> h1 (bf16)
    const int n = (wave * 4 + i) * 16 + m;
    const float bias = b1[n];
#pragma unroll
    for (int r = 0; r < 8; ++r) {
      float v = acc1[i][r] + bias;
      h1buf[mbase + r][n] = (bf16_t)(v > 0.f ? v : 0.f);
    }
  }
  __syncthreads();

  // ---------------- GEMM2: h1(16x512) @ W2(512x256) -------------------------
  v8f acc2[2] = {};
#pragma unroll 4
  for (int kt = 0; kt < NKT2; ++kt) {
    v16bf a = cat16(*(const v8bf*)&h1buf[m][kt * 32 + ksel],
                    *(const v8bf*)&h1buf[m][kt * 32 + ksel + 16]);
#pragma unroll
    for (int i = 0; i < 2; ++i) {                  // 2 N-tiles per wave
      const int nt = wave * 2 + i;
      const bf16_t* bp = W2p + (((size_t)kt * NNT2 + nt) * 32 + lane) * 16;
      v16bf bb = cat16(*(const v8bf*)bp, *(const v8bf*)(bp + 8));
      acc2[i] = __builtin_amdgcn_wmma_f32_16x16x32_bf16(
          false, a, false, bb, (short)0, acc2[i], false, false);
    }
  }
#pragma unroll
  for (int i = 0; i < 2; ++i) {                    // bias + ReLU -> h2 (bf16)
    const int n = (wave * 2 + i) * 16 + m;
    const float bias = b2[n];
#pragma unroll
    for (int r = 0; r < 8; ++r) {
      float v = acc2[i][r] + bias;
      h2buf[mbase + r][n] = (bf16_t)(v > 0.f ? v : 0.f);
    }
  }
  __syncthreads();

  // ---------------- GEMM3: h2(16x256) @ W3(256x1) ---------------------------
  if (tid < 16) {
    float s = b3[0];
    for (int k = 0; k < H2; ++k)
      s += (float)h2buf[tid][k] * W3[k];
    logits[rowbase + tid] = s;
  }
}

// ---------------------------------------------------------------------------
// Softmax over P=20 per batch row; one wave32 per row.
// ---------------------------------------------------------------------------
__global__ __launch_bounds__(32)
void softmax_kernel(const float* __restrict__ logits, float* __restrict__ out) {
  const int b = blockIdx.x;
  const int lane = threadIdx.x;
  float v = (lane < P_SZ) ? logits[b * P_SZ + lane] : -3.402823466e38f;
  float mx = v;
#pragma unroll
  for (int off = 16; off > 0; off >>= 1) mx = fmaxf(mx, __shfl_xor(mx, off));
  float e = (lane < P_SZ) ? __expf(v - mx) : 0.f;
  float s = e;
#pragma unroll
  for (int off = 16; off > 0; off >>= 1) s += __shfl_xor(s, off);
  if (lane < P_SZ) out[b * P_SZ + lane] = e / s;
}

extern "C" void kernel_launch(void* const* d_in, const int* in_sizes, int n_in,
                              void* d_out, int out_size, void* d_ws, size_t ws_size,
                              hipStream_t stream) {
  (void)in_sizes; (void)n_in; (void)out_size; (void)ws_size;
  const float* x_raw = (const float*)d_in[0];
  const int*   x_idx = (const int*)  d_in[1];
  const float* emb   = (const float*)d_in[2];
  const float* W1    = (const float*)d_in[3];
  const float* b1    = (const float*)d_in[4];
  const float* W2    = (const float*)d_in[5];
  const float* b2    = (const float*)d_in[6];
  const float* W3    = (const float*)d_in[7];
  const float* b3    = (const float*)d_in[8];
  float* out = (float*)d_out;

  char* ws = (char*)d_ws;
  bf16_t* W1p    = (bf16_t*)(ws + OFF_W1P);
  bf16_t* W2p    = (bf16_t*)(ws + OFF_W2P);
  float*  logits = (float*) (ws + OFF_LOGITS);

  const int tot1 = FIN * H1;   // 1,146,880 packed elements
  const int tot2 = H1 * H2;    //   131,072 packed elements
  pack_w_kernel<<<(tot1 + 255) / 256, 256, 0, stream>>>(W1, W1p, H1, NNT1, tot1);
  pack_w_kernel<<<(tot2 + 255) / 256, 256, 0, stream>>>(W2, W2p, H2, NNT2, tot2);

  ffnn_kernel<<<NROWS / 16, 256, 0, stream>>>(x_raw, x_idx, emb, W1p, b1,
                                              W2p, b2, W3, b3, logits);

  softmax_kernel<<<B_SZ, 32, 0, stream>>>(logits, out);
}
